// LLaMABlock_352187319179
// MI455X (gfx1250) — matrix operand order repaired and checked by reading, not compiled, not run on hardware
//
#include <hip/hip_runtime.h>
#include <cstdint>

// ---------------------------------------------------------------------------
// Types
// ---------------------------------------------------------------------------
typedef __bf16 bf16;
typedef __attribute__((ext_vector_type(16))) __bf16 v16bf;
typedef __attribute__((ext_vector_type(8)))  float  v8f;
typedef __attribute__((ext_vector_type(4)))  unsigned int uint4v;
typedef __attribute__((ext_vector_type(2)))  unsigned int uint2v;
typedef __attribute__((ext_vector_type(4)))  float float4v;

union ABfrag { v16bf v; uint4v q[2]; bf16 e[16]; };
union CFrag  { v8f  v; float e[8]; };

constexpr int kDim = 2048;
constexpr int kNH  = 16;
constexpr int kHD  = 128;   // kDim / kNH
constexpr int kFF  = 5632;
constexpr int kB   = 2;
constexpr int kS   = 2048;
constexpr int kT   = kB * kS;   // 4096 tokens

// CDNA5 async global->LDS path (ASYNCcnt), guarded so absence degrades to the
// synchronous staging path instead of breaking the compile.
#if defined(__HIP_DEVICE_COMPILE__) && defined(__gfx1250__) &&                 \
    __has_builtin(__builtin_amdgcn_global_load_async_to_lds_b128) &&           \
    __has_builtin(__builtin_amdgcn_s_wait_asynccnt)
#define USE_ASYNC 1
#else
#define USE_ASYNC 0
#endif

#if USE_ASYNC
// Parameter types per hipcc diagnostic: pointer to v4i in AS1 (global) and
// AS3 (LDS): "__attribute__((__vector_size__(4*sizeof(int)))) int __device__ *"
typedef int v4i_gcc __attribute__((vector_size(16)));
typedef __attribute__((address_space(1))) v4i_gcc* gv4i_ptr;
typedef __attribute__((address_space(3))) v4i_gcc* lv4i_ptr;

__device__ __forceinline__ void async_b128(const bf16* g, bf16* l) {
  __builtin_amdgcn_global_load_async_to_lds_b128(
      (gv4i_ptr)(uintptr_t)g, (lv4i_ptr)(uintptr_t)l,
      /*offset=*/0, /*cpol=*/0);
}
__device__ __forceinline__ void async_wait0() {
  __builtin_amdgcn_s_wait_asynccnt(0);
}
#else
__device__ __forceinline__ void async_wait0() {}
#endif

__device__ __forceinline__ v8f wmma_bf16(const ABfrag& a, const ABfrag& b, v8f c) {
  // D = A(16x32 bf16) * B(32x16 bf16) + C(16x16 f32)
  return __builtin_amdgcn_wmma_f32_16x16x32_bf16(
      /*neg_a=*/false, a.v, /*neg_b=*/false, b.v,
      /*c_mod=*/(short)0, c, /*reuse_a=*/false, /*reuse_b=*/false);
}

// ---------------------------------------------------------------------------
// fp32 -> bf16 conversion (vectorized x4)
// ---------------------------------------------------------------------------
__global__ __launch_bounds__(256) void f2bf_kernel(const float* __restrict__ in,
                                                   bf16* __restrict__ out, int n4) {
  int i = blockIdx.x * 256 + threadIdx.x;
  if (i >= n4) return;
  float4v v = ((const float4v*)in)[i];
  union { bf16 e[4]; uint2v q; } o;
  o.e[0] = (bf16)v.x; o.e[1] = (bf16)v.y; o.e[2] = (bf16)v.z; o.e[3] = (bf16)v.w;
  ((uint2v*)out)[i] = o.q;
}

// ---------------------------------------------------------------------------
// RMSNorm: one block per token row, 256 threads x 8 elems
// ---------------------------------------------------------------------------
__global__ __launch_bounds__(256) void rmsnorm_kernel(const float* __restrict__ x,
                                                      const float* __restrict__ g,
                                                      bf16* __restrict__ y) {
  const int row = blockIdx.x;
  const int tid = threadIdx.x;
  const float* xr = x + (size_t)row * kDim + tid * 8;
  float4v a = *(const float4v*)xr;
  float4v b = *(const float4v*)(xr + 4);
  float ss = a.x*a.x + a.y*a.y + a.z*a.z + a.w*a.w
           + b.x*b.x + b.y*b.y + b.z*b.z + b.w*b.w;
#pragma unroll
  for (int off = 16; off >= 1; off >>= 1) ss += __shfl_xor(ss, off, 32);
  __shared__ float red[8];
  if ((tid & 31) == 0) red[tid >> 5] = ss;
  __syncthreads();
  float tot = 0.f;
#pragma unroll
  for (int i = 0; i < 8; i++) tot += red[i];
  const float inv = rsqrtf(tot / (float)kDim + 1e-6f);
  const float* gr = g + tid * 8;
  float4v ga = *(const float4v*)gr;
  float4v gb = *(const float4v*)(gr + 4);
  union { bf16 e[8]; uint4v q; } o;
  o.e[0] = (bf16)(a.x * inv * ga.x); o.e[1] = (bf16)(a.y * inv * ga.y);
  o.e[2] = (bf16)(a.z * inv * ga.z); o.e[3] = (bf16)(a.w * inv * ga.w);
  o.e[4] = (bf16)(b.x * inv * gb.x); o.e[5] = (bf16)(b.y * inv * gb.y);
  o.e[6] = (bf16)(b.z * inv * gb.z); o.e[7] = (bf16)(b.w * inv * gb.w);
  *(uint4v*)(y + (size_t)row * kDim + tid * 8) = o.q;
}

// ---------------------------------------------------------------------------
// Tiled bf16 WMMA GEMM: C[M,N] = A[M,K] * B[K,N]
//   block tile 128x128, 8 waves (2x4), wave tile 64x32 = 4x2 WMMA tiles
//   Double-buffered LDS; A staged via async global->LDS, B via VGPR transpose.
//   EPI: 0 = store bf16, 1 = store bf16(silu), 2 = store f32 + residual
// ---------------------------------------------------------------------------
template <int EPI>
__global__ __launch_bounds__(256) void gemm_bf16_kernel(
    const bf16* __restrict__ A, const bf16* __restrict__ Bm, int K, int N,
    bf16* __restrict__ outb, float* __restrict__ outf, const float* __restrict__ res) {
  __shared__ bf16 As[2][128][40];   // row-major A tiles, padded rows (80B)
  __shared__ bf16 Bt[2][128][40];   // transposed B tiles: Bt[n][k]

  const int tid  = threadIdx.x;
  const int wave = tid >> 5, lane = tid & 31;
  const int lm = lane & 15, hh = lane >> 4;
  const int wm = wave & 1, wn = wave >> 1;
  const int bm = blockIdx.y * 128, bn = blockIdx.x * 128;

  CFrag acc[4][2];
#pragma unroll
  for (int i = 0; i < 4; i++)
#pragma unroll
    for (int j = 0; j < 2; j++)
#pragma unroll
      for (int r = 0; r < 8; r++) acc[i][j].e[r] = 0.f;

  const int arow = tid >> 1, akc = (tid & 1) * 16;   // A: 128 rows x 32 cols
  const int brow = tid >> 3, bcol = (tid & 7) * 16;  // B: 32 rows x 128 cols
  const bf16* aptr = A + (size_t)(bm + arow) * K + akc;
  const bf16* bptr = Bm + (size_t)brow * N + bn + bcol;

  auto stage_A = [&](int k0, int buf) {
#if USE_ASYNC
    async_b128(aptr + k0,     &As[buf][arow][akc]);
    async_b128(aptr + k0 + 8, &As[buf][arow][akc + 8]);
#else
    *(uint4v*)&As[buf][arow][akc]     = *(const uint4v*)(aptr + k0);
    *(uint4v*)&As[buf][arow][akc + 8] = *(const uint4v*)(aptr + k0 + 8);
#endif
  };
  auto load_B = [&](int k0, ABfrag& t) {
    const bf16* bp = bptr + (size_t)k0 * N;
    t.q[0] = *(const uint4v*)bp;
    t.q[1] = *(const uint4v*)(bp + 8);
  };
  auto store_B = [&](const ABfrag& t, int buf) {
#pragma unroll
    for (int j = 0; j < 16; j++) Bt[buf][bcol + j][brow] = t.e[j];
  };

  const int nk = K >> 5;
  int buf = 0;
  ABfrag breg;
  stage_A(0, 0);
  load_B(0, breg);
  store_B(breg, 0);
  async_wait0();
  __syncthreads();

  for (int kt = 0; kt < nk; ++kt) {
    const int nxt = buf ^ 1;
    const bool more = (kt + 1) < nk;
    if (kt + 2 < nk) {  // L2 prefetch two tiles ahead (global_prefetch_b8)
      __builtin_prefetch(aptr + (kt + 2) * 32, 0, 0);
      __builtin_prefetch(bptr + (size_t)(kt + 2) * 32 * N, 0, 0);
    }
    if (more) {
      stage_A((kt + 1) * 32, nxt);   // async: overlaps with WMMA below
      load_B((kt + 1) * 32, breg);
    }

    ABfrag af[4], bfr[2];
#pragma unroll
    for (int tm = 0; tm < 4; tm++) {
      const bf16* s = &As[buf][wm * 64 + tm * 16 + lm][0];
      af[tm].q[0] = *(const uint4v*)(s + 8 * hh);        // K = 8h..8h+7
      af[tm].q[1] = *(const uint4v*)(s + 16 + 8 * hh);   // K = 16+8h..
    }
#pragma unroll
    for (int tn = 0; tn < 2; tn++) {
      const bf16* s = &Bt[buf][wn * 32 + tn * 16 + lm][0];
      bfr[tn].q[0] = *(const uint4v*)(s + 8 * hh);       // K = 8h..8h+7
      bfr[tn].q[1] = *(const uint4v*)(s + 16 + 8 * hh);  // K = 16+8h..
    }
#pragma unroll
    for (int tm = 0; tm < 4; tm++)
#pragma unroll
      for (int tn = 0; tn < 2; tn++)
        acc[tm][tn].v = wmma_bf16(af[tm], bfr[tn], acc[tm][tn].v);

    if (more) store_B(breg, nxt);
    async_wait0();
    __syncthreads();
    buf = nxt;
  }

#pragma unroll
  for (int tm = 0; tm < 4; tm++)
#pragma unroll
    for (int tn = 0; tn < 2; tn++)
#pragma unroll
      for (int r = 0; r < 8; r++) {
        const int row = bm + wm * 64 + tm * 16 + r + 8 * hh;
        const int col = bn + wn * 32 + tn * 16 + lm;
        const size_t idx = (size_t)row * N + col;
        const float v = acc[tm][tn].e[r];
        if (EPI == 0)      outb[idx] = (bf16)v;
        else if (EPI == 1) outb[idx] = (bf16)(v / (1.f + __expf(-v)));
        else               outf[idx] = v + res[idx];
      }
}

// ---------------------------------------------------------------------------
// Causal flash attention. Block = (batch, head, 128-row q tile); 8 waves,
// each wave owns 16 q rows. K/V staged in LDS in 32-key tiles; V transposed.
// K staged via async global->LDS (overlaps the V transpose).
// qkv layout: [token][3*kDim] with q/k/v at col offsets 0/kDim/2*kDim.
// ---------------------------------------------------------------------------
__global__ __launch_bounds__(256) void attn_kernel(const bf16* __restrict__ qkv,
                                                   bf16* __restrict__ out) {
  constexpr int QTILES = kS / 128;  // 16
  const int qt = blockIdx.x % QTILES;
  const int h  = (blockIdx.x / QTILES) % kNH;
  const int b  = blockIdx.x / (QTILES * kNH);

  __shared__ bf16 Ks[32][136];     // K tile, row-major  (keys x hd)
  __shared__ bf16 Vt[128][40];     // V tile, transposed (hd x keys)
  __shared__ bf16 Pw[8][16][40];   // per-wave P scratch (16 q x 32 keys)

  const int tid  = threadIdx.x;
  const int wave = tid >> 5, lane = tid & 31;
  const int lm = lane & 15, hl = lane >> 4;
  const int qrow0 = qt * 128 + wave * 16;

  // Preload Q fragments: hd = 4 chunks of K=32
  ABfrag Qf[4];
  {
    const bf16* qp = qkv + (size_t)(b * kS + qrow0 + lm) * (3 * kDim) + h * kHD;
#pragma unroll
    for (int kk = 0; kk < 4; kk++) {
      Qf[kk].q[0] = *(const uint4v*)(qp + kk * 32 + 8 * hl);
      Qf[kk].q[1] = *(const uint4v*)(qp + kk * 32 + 16 + 8 * hl);
    }
  }

  CFrag O[8];
  float mrun[8], lrun[8];
#pragma unroll
  for (int o = 0; o < 8; o++)
#pragma unroll
    for (int r = 0; r < 8; r++) O[o].e[r] = 0.f;
#pragma unroll
  for (int r = 0; r < 8; r++) { mrun[r] = -1e30f; lrun[r] = 0.f; }

  const int stg_row = tid >> 3;        // 0..31 (key within tile)
  const int stg_cb  = (tid & 7) * 16;  // hd column block
  const int nkb = qt * 4 + 4;          // causal: keys up to q-tile end

  for (int kb = 0; kb < nkb; kb++) {
    // --- stage K (async, row-major) and V (transposed via VGPR) in LDS ---
    const size_t trow = (size_t)(b * kS + kb * 32 + stg_row) * (3 * kDim);
    const bf16* kp = qkv + trow + kDim + h * kHD + stg_cb;
#if USE_ASYNC
    async_b128(kp,     &Ks[stg_row][stg_cb]);
    async_b128(kp + 8, &Ks[stg_row][stg_cb + 8]);
#else
    *(uint4v*)&Ks[stg_row][stg_cb]     = *(const uint4v*)kp;
    *(uint4v*)&Ks[stg_row][stg_cb + 8] = *(const uint4v*)(kp + 8);
#endif
    const bf16* vp = qkv + trow + 2 * kDim + h * kHD + stg_cb;
    ABfrag vtmp;
    vtmp.q[0] = *(const uint4v*)vp; vtmp.q[1] = *(const uint4v*)(vp + 8);
#pragma unroll
    for (int j = 0; j < 16; j++) Vt[stg_cb + j][stg_row] = vtmp.e[j];
    async_wait0();
    __syncthreads();

    // --- scores S = Q @ K^T : two 16x16 tiles over 32 keys ---
    CFrag S[2];
#pragma unroll
    for (int nt = 0; nt < 2; nt++) {
#pragma unroll
      for (int r = 0; r < 8; r++) S[nt].e[r] = 0.f;
      const bf16* ks = &Ks[nt * 16 + lm][0];
#pragma unroll
      for (int kk = 0; kk < 4; kk++) {
        ABfrag kf;
        kf.q[0] = *(const uint4v*)(ks + kk * 32 + 8 * hl);
        kf.q[1] = *(const uint4v*)(ks + kk * 32 + 16 + 8 * hl);
        S[nt].v = wmma_bf16(Qf[kk], kf, S[nt].v);
      }
    }

    // --- scale, causal mask, online softmax ---
    const float sc = 0.08838834764831845f;  // 1/sqrt(128)
#pragma unroll
    for (int r = 0; r < 8; r++) {
      const int sq = qrow0 + r + 8 * hl;
#pragma unroll
      for (int nt = 0; nt < 2; nt++) {
        const int sk = kb * 32 + nt * 16 + lm;
        const float v = S[nt].e[r] * sc;
        S[nt].e[r] = (sk <= sq) ? v : -1e30f;
      }
      float mx = fmaxf(S[0].e[r], S[1].e[r]);
#pragma unroll
      for (int off = 8; off >= 1; off >>= 1) mx = fmaxf(mx, __shfl_xor(mx, off, 16));
      const float nm = fmaxf(mrun[r], mx);
      const float al = __expf(mrun[r] - nm);
      mrun[r] = nm;
      float ps = 0.f;
#pragma unroll
      for (int nt = 0; nt < 2; nt++) {
        const float pv = __expf(S[nt].e[r] - nm);
        ps += pv;
        Pw[wave][r + 8 * hl][nt * 16 + lm] = (bf16)pv;  // C-layout -> LDS
      }
#pragma unroll
      for (int off = 8; off >= 1; off >>= 1) ps += __shfl_xor(ps, off, 16);
      lrun[r] = lrun[r] * al + ps;
#pragma unroll
      for (int o = 0; o < 8; o++) O[o].e[r] *= al;
    }

    // --- O += P @ V (LDS ops in-order within a wave; no barrier needed) ---
    ABfrag pf;
    pf.q[0] = *(const uint4v*)&Pw[wave][lm][8 * hl];
    pf.q[1] = *(const uint4v*)&Pw[wave][lm][16 + 8 * hl];
#pragma unroll
    for (int o = 0; o < 8; o++) {
      ABfrag vf;
      const bf16* vs = &Vt[o * 16 + lm][0];
      vf.q[0] = *(const uint4v*)(vs + 8 * hl);
      vf.q[1] = *(const uint4v*)(vs + 16 + 8 * hl);
      O[o].v = wmma_bf16(pf, vf, O[o].v);
    }
    __syncthreads();
  }

  // --- normalize + store (out layout [token][h*128 + d]) ---
  bf16* op = out + (size_t)(b * kS + qrow0) * kDim + h * kHD;
#pragma unroll
  for (int r = 0; r < 8; r++) {
    const float inv = 1.f / lrun[r];
    const size_t ro = (size_t)(r + 8 * hl) * kDim;
#pragma unroll
    for (int o = 0; o < 8; o++) op[ro + o * 16 + lm] = (bf16)(O[o].e[r] * inv);
  }
}

// ---------------------------------------------------------------------------
// SwiGLU elementwise: a = a * b (a already silu'd), vectorized x4, in place
// ---------------------------------------------------------------------------
__global__ __launch_bounds__(256) void swiglu_mul_kernel(bf16* __restrict__ a,
                                                         const bf16* __restrict__ b,
                                                         int n4) {
  int i = blockIdx.x * 256 + threadIdx.x;
  if (i >= n4) return;
  union U { bf16 e[4]; uint2v q; } ua, ub;
  ua.q = ((uint2v*)a)[i];
  ub.q = ((const uint2v*)b)[i];
#pragma unroll
  for (int j = 0; j < 4; j++) ua.e[j] = (bf16)((float)ua.e[j] * (float)ub.e[j]);
  ((uint2v*)a)[i] = ua.q;
}

// ---------------------------------------------------------------------------
// Launch
// ---------------------------------------------------------------------------
extern "C" void kernel_launch(void* const* d_in, const int* in_sizes, int n_in,
                              void* d_out, int out_size, void* d_ws, size_t ws_size,
                              hipStream_t stream) {
  (void)in_sizes; (void)n_in; (void)out_size; (void)ws_size;
  const float* x    = (const float*)d_in[0];
  const float* wqkv = (const float*)d_in[1];
  const float* wout = (const float*)d_in[2];
  const float* g1   = (const float*)d_in[3];
  const float* g2   = (const float*)d_in[4];
  const float* w1   = (const float*)d_in[5];
  const float* w3   = (const float*)d_in[6];
  const float* w2   = (const float*)d_in[7];
  float* outp = (float*)d_out;

  char* p = (char*)d_ws;
  auto alloc_b = [&](size_t bytes) -> void* {
    void* r = (void*)p;
    p += (bytes + 255) & ~(size_t)255;
    return r;
  };
  bf16*  wqkvb = (bf16*)alloc_b((size_t)kDim * 3 * kDim * sizeof(bf16));
  bf16*  woutb = (bf16*)alloc_b((size_t)kDim * kDim * sizeof(bf16));
  bf16*  w1b   = (bf16*)alloc_b((size_t)kDim * kFF * sizeof(bf16));
  bf16*  w3b   = (bf16*)alloc_b((size_t)kDim * kFF * sizeof(bf16));
  bf16*  w2b   = (bf16*)alloc_b((size_t)kFF * kDim * sizeof(bf16));
  bf16*  xn    = (bf16*)alloc_b((size_t)kT * kDim * sizeof(bf16));
  bf16*  qkvb  = (bf16*)alloc_b((size_t)kT * 3 * kDim * sizeof(bf16));
  bf16*  aob   = (bf16*)alloc_b((size_t)kT * kDim * sizeof(bf16));
  float* x1    = (float*)alloc_b((size_t)kT * kDim * sizeof(float));
  bf16*  s1    = (bf16*)alloc_b((size_t)kT * kFF * sizeof(bf16));
  bf16*  h3    = (bf16*)alloc_b((size_t)kT * kFF * sizeof(bf16));

  auto cvt = [&](const float* src, bf16* dst, size_t n) {
    int n4 = (int)(n / 4);
    f2bf_kernel<<<(n4 + 255) / 256, 256, 0, stream>>>(src, dst, n4);
  };
  cvt(wqkv, wqkvb, (size_t)kDim * 3 * kDim);
  cvt(wout, woutb, (size_t)kDim * kDim);
  cvt(w1,   w1b,   (size_t)kDim * kFF);
  cvt(w3,   w3b,   (size_t)kDim * kFF);
  cvt(w2,   w2b,   (size_t)kFF * kDim);

  // x1 = x + attn(rmsnorm(x, g1))
  rmsnorm_kernel<<<kT, 256, 0, stream>>>(x, g1, xn);
  gemm_bf16_kernel<0><<<dim3(3 * kDim / 128, kT / 128), 256, 0, stream>>>(
      xn, wqkvb, kDim, 3 * kDim, qkvb, nullptr, nullptr);
  attn_kernel<<<kB * kNH * (kS / 128), 256, 0, stream>>>(qkvb, aob);
  gemm_bf16_kernel<2><<<dim3(kDim / 128, kT / 128), 256, 0, stream>>>(
      aob, woutb, kDim, kDim, nullptr, x1, x);

  // out = x1 + swiglu(rmsnorm(x1, g2))
  rmsnorm_kernel<<<kT, 256, 0, stream>>>(x1, g2, xn);
  gemm_bf16_kernel<1><<<dim3(kFF / 128, kT / 128), 256, 0, stream>>>(
      xn, w1b, kDim, kFF, s1, nullptr, nullptr);
  gemm_bf16_kernel<0><<<dim3(kFF / 128, kT / 128), 256, 0, stream>>>(
      xn, w3b, kDim, kFF, h3, nullptr, nullptr);
  {
    int n4 = (int)((size_t)kT * kFF / 4);
    swiglu_mul_kernel<<<(n4 + 255) / 256, 256, 0, stream>>>(s1, h3, n4);
  }
  gemm_bf16_kernel<2><<<dim3(kDim / 128, kT / 128), 256, 0, stream>>>(
      s1, w2b, kFF, kDim, nullptr, outp, x1);
}
